// EulerIntegrator_40080634806916
// MI455X (gfx1250) — matrix operand order repaired and checked by reading, not compiled, not run on hardware
//
#include <hip/hip_runtime.h>
#include <math.h>

typedef __attribute__((ext_vector_type(16))) __bf16 v16bf;
typedef __attribute__((ext_vector_type(8)))  __bf16 v8bf;
typedef __attribute__((ext_vector_type(8)))  float  v8f;

#define B_TOT 4096
#define D_DIM 256
#define H_DIM 1024
#define BM    16            // batch rows per workgroup
#define SVB_PAD 8           // bf16 pad -> row stride 528B  (bank spread, 16B aligned)
#define SUV_PAD 8           // bf16 pad -> row stride 2064B (bank spread, 16B aligned)
#define PI_F  3.14159265358979323846f
#define DT_F  0.01f

// one-shot: convert U and W (f32) to bf16 in workspace (native v_cvt path)
__global__ void cvt_bf16_kernel(const float* __restrict__ U, const float* __restrict__ W,
                                __bf16* __restrict__ Ubf, __bf16* __restrict__ Wbf,
                                int n) {
    int i = blockIdx.x * blockDim.x + threadIdx.x;
    if (i < n) {
        Ubf[i] = (__bf16)U[i];
        Wbf[i] = (__bf16)W[i];
    }
}

__global__ void __launch_bounds__(256)
euler_wmma_kernel(const float* __restrict__ gx, const float* __restrict__ gv,
                  const float* __restrict__ gf,
                  const __bf16* __restrict__ Ubf,
                  const __bf16* __restrict__ Wbf,
                  const int* __restrict__ d_steps,
                  float* __restrict__ out) {
    // bf16 shadow of v (GEMM1 A operand, shared across waves) + squared-uv buffer
    __shared__ __align__(16) __bf16 svb[BM][D_DIM + SVB_PAD];
    __shared__ __align__(16) __bf16 suv[BM][H_DIM + SUV_PAD];

    const int tid  = threadIdx.x;
    const int wave = tid >> 5;              // 0..7
    const int lane = tid & 31;
    const int half = lane >> 4;             // 0|1 (WMMA half-wave)
    const int l16  = lane & 15;
    const int rb   = blockIdx.x * BM;       // batch-row base of this tile
    const int steps = d_steps[0];
    const int dbase = 32 * wave;            // this wave's d-slice for GEMM2/update

    // cooperative fill of the bf16 v shadow (coalesced)
    for (int i = 0; i < (BM * D_DIM) / 256; ++i) {
        int idx = tid + i * 256;
        int r = idx >> 8, c = idx & 255;
        svb[r][c] = (__bf16)gv[(size_t)(rb + r) * D_DIM + c];
    }

    // x, v, force held in registers in the exact WMMA C/D lane mapping:
    // tile n in {0,1}: d = dbase + 16n + l16, row m = r + 8*half
    float xreg[2][8], vreg[2][8], freg[2][8];
    #pragma unroll
    for (int n = 0; n < 2; ++n) {
        int d = dbase + 16 * n + l16;
        #pragma unroll
        for (int r = 0; r < 8; ++r) {
            int m = r + 8 * half;
            xreg[n][r] = gx[(size_t)(rb + m) * D_DIM + d];
            vreg[n][r] = gv[(size_t)(rb + m) * D_DIM + d];
            freg[n][r] = gf[(size_t)(rb + m) * D_DIM + d];
        }
    }
    __syncthreads();

    const float twopi = 2.0f * PI_F;

    for (int s = 0; s < steps; ++s) {
        // ============ GEMM1: uv = v @ U^T ; suv = uv*uv (bf16) ============
        // A fragments depend only on kb -> hoist all 8 (pure ds_load_b128 path)
        v16bf afrag[8];
        #pragma unroll
        for (int kb = 0; kb < 8; ++kb) {
            const int c = kb * 32 + (half ? 16 : 0);
            v8bf lo = *(const v8bf*)&svb[l16][c];
            v8bf hi = *(const v8bf*)&svb[l16][c + 8];
            afrag[kb] = __builtin_shufflevector(lo, hi,
                          0,1,2,3,4,5,6,7,8,9,10,11,12,13,14,15);
        }
        const int hwave = 128 * wave;       // this wave's h-slice
        // process h-tiles in pairs: two live accumulators, shared A-fragment
        #pragma unroll 1
        for (int n2 = 0; n2 < 4; ++n2) {
            const int hb0 = hwave + 32 * n2;
            const int hb1 = hb0 + 16;
            v8f acc0 = {}, acc1 = {};
            #pragma unroll
            for (int kb = 0; kb < 8; ++kb) {
                const int kstart = kb * 32 + (half ? 16 : 0);
                v16bf b0 = *(const v16bf*)(Ubf + (size_t)(hb0 + l16) * D_DIM + kstart);
                v16bf b1 = *(const v16bf*)(Ubf + (size_t)(hb1 + l16) * D_DIM + kstart);
                acc0 = __builtin_amdgcn_wmma_f32_16x16x32_bf16(
                        false, afrag[kb], false, b0, (short)0, acc0, false, false);
                acc1 = __builtin_amdgcn_wmma_f32_16x16x32_bf16(
                        false, afrag[kb], false, b1, (short)0, acc1, false, false);
            }
            #pragma unroll
            for (int r = 0; r < 8; ++r) {
                float q0 = acc0[r];
                float q1 = acc1[r];
                suv[r + 8 * half][hb0 + l16] = (__bf16)(q0 * q0);
                suv[r + 8 * half][hb1 + l16] = (__bf16)(q1 * q1);
            }
        }
        __syncthreads();   // suv complete (all h); v-shadow reads done

        // ============ GEMM2: c = suv @ W^T over this wave's 32-wide d slice ============
        v8f cacc0 = {}, cacc1 = {};
        #pragma unroll 2
        for (int kb = 0; kb < 32; ++kb) {
            const int c = kb * 32 + (half ? 16 : 0);
            v8bf lo = *(const v8bf*)&suv[l16][c];
            v8bf hi = *(const v8bf*)&suv[l16][c + 8];
            v16bf a = __builtin_shufflevector(lo, hi,
                        0,1,2,3,4,5,6,7,8,9,10,11,12,13,14,15);
            const int kstart = kb * 32 + (half ? 16 : 0);
            v16bf b0 = *(const v16bf*)(Wbf + (size_t)(dbase      + l16) * H_DIM + kstart);
            v16bf b1 = *(const v16bf*)(Wbf + (size_t)(dbase + 16 + l16) * H_DIM + kstart);
            cacc0 = __builtin_amdgcn_wmma_f32_16x16x32_bf16(
                        false, a, false, b0, (short)0, cacc0, false, false);
            cacc1 = __builtin_amdgcn_wmma_f32_16x16x32_bf16(
                        false, a, false, b1, (short)0, cacc1, false, false);
        }

        // ============ elementwise Euler update (all f32, register state) ============
        #pragma unroll
        for (int n = 0; n < 2; ++n) {
            const int d = dbase + 16 * n + l16;
            #pragma unroll
            for (int r = 0; r < 8; ++r) {
                const int m = r + 8 * half;
                float cval = (n == 0) ? cacc0[r] : cacc1[r];
                float vold = vreg[n][r];
                float xn = xreg[n][r] + DT_F * vold + PI_F;
                xn = xn - twopi * floorf(xn / twopi);     // jnp.mod semantics
                xreg[n][r] = xn - PI_F;
                float vnew = vold + DT_F * (freg[n][r] - cval);
                vreg[n][r] = vnew;
                svb[m][d] = (__bf16)vnew;                 // refresh bf16 shadow
            }
        }
        __syncthreads();   // v-shadow updated before next step's GEMM1 reads
    }

    // write cx then cv (concatenated flat in return order)
    #pragma unroll
    for (int n = 0; n < 2; ++n) {
        const int d = dbase + 16 * n + l16;
        #pragma unroll
        for (int r = 0; r < 8; ++r) {
            const int m = r + 8 * half;
            out[(size_t)(rb + m) * D_DIM + d] = xreg[n][r];
            out[(size_t)B_TOT * D_DIM + (size_t)(rb + m) * D_DIM + d] = vreg[n][r];
        }
    }
}

extern "C" void kernel_launch(void* const* d_in, const int* in_sizes, int n_in,
                              void* d_out, int out_size, void* d_ws, size_t ws_size,
                              hipStream_t stream) {
    const float* gx = (const float*)d_in[0];
    const float* gv = (const float*)d_in[1];
    const float* gf = (const float*)d_in[2];
    const float* gU = (const float*)d_in[3];
    const float* gW = (const float*)d_in[4];
    const int* gsteps = (const int*)d_in[5];

    __bf16* Ubf = (__bf16*)d_ws;                           // 1024*256 bf16
    __bf16* Wbf = Ubf + (size_t)H_DIM * D_DIM;             // 256*1024 bf16

    const int n = H_DIM * D_DIM;
    cvt_bf16_kernel<<<(n + 255) / 256, 256, 0, stream>>>(gU, gW, Ubf, Wbf, n);

    euler_wmma_kernel<<<B_TOT / BM, 256, 0, stream>>>(
        gx, gv, gf, Ubf, Wbf, gsteps, (float*)d_out);
}
